// PDUMemoryCell_28673201668585
// MI455X (gfx1250) — matrix operand order repaired and checked by reading, not compiled, not run on hardware
//
#include <hip/hip_runtime.h>

// ---------------------------------------------------------------------------
// Types
// ---------------------------------------------------------------------------
typedef __bf16 bf16_t;
typedef bf16_t v16bf __attribute__((ext_vector_type(16)));
typedef bf16_t v8bf  __attribute__((ext_vector_type(8)));
typedef bf16_t v4bf  __attribute__((ext_vector_type(4)));
typedef float  v8f   __attribute__((ext_vector_type(8)));
typedef float  v4f   __attribute__((ext_vector_type(4)));

#define CHUNK_L 128
#define EPS_V   1e-6f

// Async global->LDS 16B copy (CDNA5): per-lane LDS offset in a VGPR, global
// address in a VGPR pair, tracked with ASYNCcnt.
#define ASYNC_G2L_B128(ldsOff, gptr)                                          \
    asm volatile("global_load_async_to_lds_b128 %0, %1, off"                  \
                 :: "v"(ldsOff), "v"((unsigned long long)(gptr)) : "memory")
#define WAIT_ASYNCCNT0() asm volatile("s_wait_asynccnt 0x0" ::: "memory")

// ---------------------------------------------------------------------------
// fp32 -> bf16 conversion (vectorized 4-wide, grid-stride)
// ---------------------------------------------------------------------------
__global__ __launch_bounds__(256) void cvt_f32_to_bf16(const float* __restrict__ src,
                                                       bf16_t* __restrict__ dst,
                                                       long long n4) {
    long long i = (long long)blockIdx.x * blockDim.x + threadIdx.x;
    long long stride = (long long)gridDim.x * blockDim.x;
    for (; i < n4; i += stride) {
        v4f f = *(const v4f*)(src + 4 * i);
        v4bf o;
        o[0] = (bf16_t)f[0];
        o[1] = (bf16_t)f[1];
        o[2] = (bf16_t)f[2];
        o[3] = (bf16_t)f[3];
        *(v4bf*)(dst + 4 * i) = o;
    }
}

// ---------------------------------------------------------------------------
// bf16 WMMA GEMM:  C(MxN) = A(MxK) @ B(KxN) + bias[N], optional ReLU.
//
// Block tile 128x256, K-step 32, 256 threads = 8 waves (wave32).
// Wave grid 2(m) x 4(n); each wave owns a 64x64 tile = 4x4 WMMA accumulators.
// Double-buffered LDS; A staged via global_load_async_to_lds_b128 (ASYNCcnt),
// B register-staged with compile-time immediate offsets (N, K template args).
//
// LDS fragment order (contiguous k, same for A and B):
//   A frag (16x32): lane l -> row M=l%16; elem j -> K=(l/16)*16+j
//   B frag (32x16): lane l -> col N=l%16; elem j -> K=(l/16)*16+j
// ---------------------------------------------------------------------------
template <int N, int K, bool RELU, bool OUT_BF16>
__global__ __launch_bounds__(256) void gemm_bf16_wmma(
    const bf16_t* __restrict__ A, const bf16_t* __restrict__ Bm,
    const float* __restrict__ bias, void* __restrict__ Cout, int M)
{
    __shared__ alignas(32) bf16_t aSh[2][8][32][16];    // 16 KB (2 bufs)
    __shared__ alignas(32) bf16_t bSh[2][16][32][16];   // 32 KB (2 bufs)

    const int tid  = threadIdx.x;
    const int lane = tid & 31;
    const int wave = tid >> 5;
    const int wm   = wave & 1;        // 2 m positions (64 rows each)
    const int wn   = wave >> 1;       // 4 n positions (64 cols each)
    const int l16  = lane & 15;
    const int hf   = lane >> 4;

    const int blockM = blockIdx.y * 128;
    const int blockN = blockIdx.x * 256;

    v8f acc[4][4];
    #pragma unroll
    for (int i = 0; i < 4; ++i)
        #pragma unroll
        for (int j = 0; j < 4; ++j)
            #pragma unroll
            for (int r = 0; r < 8; ++r)
                acc[i][j][r] = 0.0f;

    // Loader roles: wave w stages A tile w and B tiles {w, w+8}.
    const bf16_t* aRow = A + (long long)(blockM + wave * 16 + l16) * K + hf * 16;
    const unsigned aLds0 = (unsigned)(unsigned long long)&aSh[0][wave][lane][0];
    const unsigned aLds1 = (unsigned)(unsigned long long)&aSh[1][wave][lane][0];
    const bf16_t* bPtr = Bm + (long long)(hf * 16) * N + (blockN + wave * 16 + l16);

    // ---- prologue: stage k0 = 0 into buffer 0
    {
        ASYNC_G2L_B128(aLds0,       aRow);
        ASYNC_G2L_B128(aLds0 + 16u, aRow + 8);
        v8bf r0, r1, r2, r3;
        #pragma unroll
        for (int j = 0; j < 8; ++j) {
            r0[j] = bPtr[j * N];              // tile 'wave',   K = hf*16 + j
            r1[j] = bPtr[(j + 8) * N];        // tile 'wave',   K = hf*16 + 8 + j
            r2[j] = bPtr[j * N + 128];        // tile 'wave+8'
            r3[j] = bPtr[(j + 8) * N + 128];
        }
        *(v8bf*)&bSh[0][wave][lane][0]     = r0;
        *(v8bf*)&bSh[0][wave][lane][8]     = r1;
        *(v8bf*)&bSh[0][wave + 8][lane][0] = r2;
        *(v8bf*)&bSh[0][wave + 8][lane][8] = r3;
        WAIT_ASYNCCNT0();
    }
    __syncthreads();

    for (int k0 = 0; k0 < K; k0 += 32) {
        const int cur = (k0 >> 5) & 1;
        const int nxt = cur ^ 1;
        const bool more = (k0 + 32) < K;

        v8bf r0, r1, r2, r3;
        if (more) {
            // Prefetch next K-slab: A async to LDS (other buffer), B to regs.
            const bf16_t* aNext = aRow + (k0 + 32);
            const unsigned aL = nxt ? aLds1 : aLds0;
            ASYNC_G2L_B128(aL,       aNext);
            ASYNC_G2L_B128(aL + 16u, aNext + 8);
            const bf16_t* bp = bPtr + (long long)(k0 + 32) * N;
            #pragma unroll
            for (int j = 0; j < 8; ++j) {
                r0[j] = bp[j * N];
                r1[j] = bp[(j + 8) * N];
                r2[j] = bp[j * N + 128];
                r3[j] = bp[(j + 8) * N + 128];
            }
        }

        // 16 WMMAs on the current buffer
        #pragma unroll
        for (int mt = 0; mt < 4; ++mt) {
            v16bf va = *(const v16bf*)&aSh[cur][wm * 4 + mt][lane][0];
            #pragma unroll
            for (int nt = 0; nt < 4; ++nt) {
                v16bf vb = *(const v16bf*)&bSh[cur][wn * 4 + nt][lane][0];
                acc[mt][nt] = __builtin_amdgcn_wmma_f32_16x16x32_bf16(
                    /*neg_a=*/false, va, /*neg_b=*/false, vb,
                    /*c_mod=*/(short)0, acc[mt][nt],
                    /*reuse_a=*/false, /*reuse_b=*/false);
            }
        }

        if (more) {
            *(v8bf*)&bSh[nxt][wave][lane][0]     = r0;
            *(v8bf*)&bSh[nxt][wave][lane][8]     = r1;
            *(v8bf*)&bSh[nxt][wave + 8][lane][0] = r2;
            *(v8bf*)&bSh[nxt][wave + 8][lane][8] = r3;
            WAIT_ASYNCCNT0();
        }
        __syncthreads();
    }

    // Epilogue per C/D layout: VGPR r -> M = (lane>=16 ? 8 : 0) + r, lane%16 -> N
    #pragma unroll
    for (int mt = 0; mt < 4; ++mt) {
        #pragma unroll
        for (int nt = 0; nt < 4; ++nt) {
            const int col = blockN + (wn * 4 + nt) * 16 + l16;
            const float bv = bias[col];
            #pragma unroll
            for (int r = 0; r < 8; ++r) {
                const int row = blockM + (wm * 4 + mt) * 16 + hf * 8 + r;
                float v = acc[mt][nt][r] + bv;
                if (RELU) v = fmaxf(v, 0.0f);
                const long long idx = (long long)row * N + col;
                if (OUT_BF16) ((bf16_t*)Cout)[idx] = (bf16_t)v;
                else          ((float*)Cout)[idx]  = v;
            }
        }
    }
}

// ---------------------------------------------------------------------------
// In-chunk sequential complex scan + read computation.
// One thread per (batch, chunk, d). 128 sequential steps, coalesced over d.
// gates row layout: [rq | rps | rs | upd | dec | ph], each D wide.
// ---------------------------------------------------------------------------
__global__ __launch_bounds__(256) void pdu_scan_read(
    const float* __restrict__ gates, bf16_t* __restrict__ readout, int D)
{
    const long long idx = (long long)blockIdx.x * blockDim.x + threadIdx.x;
    const int d = (int)(idx % D);
    const long long chunkIdx = idx / D;            // over B*C
    const long long row0 = chunkIdx * CHUNK_L;
    const long long gstride = 6LL * D;

    const float* g = gates + row0 * gstride + d;
    bf16_t* out = readout + row0 * D + d;

    float mr = 0.0f, mi = 0.0f;
    for (int t = 0; t < CHUNK_L; ++t) {
        const float rq  = g[0];
        const float rps = g[(long long)D];
        const float rsx = g[2LL * D];
        const float upd = g[3LL * D];
        const float dec = g[4LL * D];
        const float ph  = g[5LL * D];

        const float rs   = (rsx > 20.0f) ? rsx : log1pf(__expf(rsx));
        const float amag = __expf(-fmaxf(dec, 0.0f));
        float sph, cph;
        __sincosf(ph, &sph, &cph);
        const float nmr = amag * (cph * mr - sph * mi) + upd;
        const float nmi = amag * (cph * mi + sph * mr);
        mr = nmr; mi = nmi;

        float sps, cps;
        __sincosf(rps, &sps, &cps);
        const float r = mr * cps - mi * sps;
        const float mag = __expf(rs * __logf(fabsf(r) + EPS_V));
        const float sgn = (r > 0.0f) ? 1.0f : ((r < 0.0f) ? -1.0f : 0.0f);
        *out = (bf16_t)(rq * sgn * mag);

        g += gstride;
        out += D;
    }
}

// ---------------------------------------------------------------------------
// LayerNorm(proj)*g + b + x   (one block of 256 per row, D = 1024)
// ---------------------------------------------------------------------------
__global__ __launch_bounds__(256) void layernorm_residual(
    const float* __restrict__ proj, const float* __restrict__ x,
    const float* __restrict__ lng, const float* __restrict__ lnb,
    float* __restrict__ out, int D)
{
    const long long base = (long long)blockIdx.x * D;
    float v[4];
    float sum = 0.0f, sq = 0.0f;
    #pragma unroll
    for (int i = 0; i < 4; ++i) {
        v[i] = proj[base + threadIdx.x + i * 256];
        sum += v[i];
        sq  += v[i] * v[i];
    }
    #pragma unroll
    for (int off = 16; off > 0; off >>= 1) {
        sum += __shfl_down(sum, off, 32);
        sq  += __shfl_down(sq,  off, 32);
    }
    __shared__ float ssum[8], ssq[8];
    __shared__ float mu_s, rstd_s;
    if ((threadIdx.x & 31) == 0) {
        ssum[threadIdx.x >> 5] = sum;
        ssq[threadIdx.x >> 5]  = sq;
    }
    __syncthreads();
    if (threadIdx.x == 0) {
        float ts = 0.0f, tq = 0.0f;
        #pragma unroll
        for (int i = 0; i < 8; ++i) { ts += ssum[i]; tq += ssq[i]; }
        const float mu  = ts / (float)D;
        const float var = tq / (float)D - mu * mu;
        mu_s = mu;
        rstd_s = rsqrtf(var + 1e-5f);
    }
    __syncthreads();
    const float mu = mu_s, rstd = rstd_s;
    #pragma unroll
    for (int i = 0; i < 4; ++i) {
        const int c = threadIdx.x + i * 256;
        out[base + c] = x[base + c] + (v[i] - mu) * rstd * lng[c] + lnb[c];
    }
}

// ---------------------------------------------------------------------------
// Launch
// ---------------------------------------------------------------------------
extern "C" void kernel_launch(void* const* d_in, const int* in_sizes, int n_in,
                              void* d_out, int out_size, void* d_ws, size_t ws_size,
                              hipStream_t stream) {
    const float* x    = (const float*)d_in[0];
    const float* W_h  = (const float*)d_in[1];
    const float* b_h  = (const float*)d_in[2];
    const float* W_o  = (const float*)d_in[3];
    const float* b_o  = (const float*)d_in[4];
    const float* W_r  = (const float*)d_in[5];
    const float* b_r  = (const float*)d_in[6];
    const float* ln_g = (const float*)d_in[7];
    const float* ln_b = (const float*)d_in[8];

    const int Bb = 4, T = 4096, Dm = 1024;
    const int G  = 6 * 1024;            // gate width
    const int M  = Bb * T;              // 16384 rows

    // Workspace carve-out (all regions fully written before read)
    char* ws = (char*)d_ws;
    size_t off = 0;
    auto carve = [&](size_t bytes) -> char* {
        char* p = ws + off;
        off += (bytes + 255) & ~(size_t)255;
        return p;
    };
    bf16_t* xb    = (bf16_t*)carve((size_t)M * Dm * 2);
    bf16_t* Whb   = (bf16_t*)carve((size_t)Dm * Dm * 2);
    bf16_t* Wob   = (bf16_t*)carve((size_t)Dm * G * 2);
    bf16_t* Wrb   = (bf16_t*)carve((size_t)Dm * Dm * 2);
    bf16_t* hidb  = (bf16_t*)carve((size_t)M * Dm * 2);
    float*  gatf  = (float*)carve((size_t)M * G * 4);
    bf16_t* rdb   = (bf16_t*)carve((size_t)M * Dm * 2);
    float*  projf = (float*)carve((size_t)M * Dm * 4);

    auto cvt = [&](const float* s, bf16_t* dptr, long long n) {
        const long long n4 = n / 4;
        long long nb = (n4 + 255) / 256;
        const int blocks = (nb > 4096) ? 4096 : (int)nb;
        cvt_f32_to_bf16<<<blocks, 256, 0, stream>>>(s, dptr, n4);
    };
    cvt(x,   xb,  (long long)M * Dm);
    cvt(W_h, Whb, (long long)Dm * Dm);
    cvt(W_o, Wob, (long long)Dm * G);
    cvt(W_r, Wrb, (long long)Dm * Dm);

    const dim3 blk(256);
    // hidden = relu(x @ W_h + b_h)  -> bf16
    gemm_bf16_wmma<1024, 1024, true, true>
        <<<dim3(Dm / 256, M / 128), blk, 0, stream>>>(xb, Whb, b_h, hidb, M);
    // gates = hidden @ W_o + b_o    -> fp32
    gemm_bf16_wmma<6144, 1024, false, false>
        <<<dim3(G / 256, M / 128), blk, 0, stream>>>(hidb, Wob, b_o, gatf, M);
    // scan + read                   -> bf16
    const int scanThreads = Bb * (T / CHUNK_L) * Dm;   // 131072
    pdu_scan_read<<<scanThreads / 256, 256, 0, stream>>>(gatf, rdb, Dm);
    // proj = read @ W_r + b_r       -> fp32
    gemm_bf16_wmma<1024, 1024, false, false>
        <<<dim3(Dm / 256, M / 128), blk, 0, stream>>>(rdb, Wrb, b_r, projf, M);
    // out = x + LN(proj)*g + b
    layernorm_residual<<<M, 256, 0, stream>>>(projf, x, ln_g, ln_b, (float*)d_out, Dm);
}
